// SphereViT_70368744178022
// MI455X (gfx1250) — matrix operand order repaired and checked by reading, not compiled, not run on hardware
//
#include <hip/hip_runtime.h>

// ---------------------------------------------------------------------------
// SphereViT forward for MI455X (gfx1250, wave32).
// All GEMMs: bf16 WMMA 16x16x32, fp32 accumulate. LN/softmax/statistics fp32.
// GEMM tiles staged LDS via Tensor Data Mover (double-buffered), weights
// pre-transposed to [N,K] bf16 once per launch (bandwidth-trivial at 23.3TB/s).
// ---------------------------------------------------------------------------

typedef unsigned short u16;
typedef unsigned int   u32;
typedef unsigned long long u64;
typedef __attribute__((ext_vector_type(4)))  u32    v4u;
typedef __attribute__((ext_vector_type(4)))  int    v4i;
typedef __attribute__((ext_vector_type(8)))  int    v8i;
typedef __attribute__((ext_vector_type(4)))  float  v4f;
typedef __attribute__((ext_vector_type(8)))  float  v8f;
typedef __attribute__((ext_vector_type(16))) __bf16 v16bf;

struct FragU { v4u a, b; };   // 32 bytes, bit-cast to v16bf

#define DEV __device__ __forceinline__

#if __has_builtin(__builtin_amdgcn_tensor_load_to_lds) && \
    __has_builtin(__builtin_amdgcn_s_wait_tensorcnt)
#define USE_TDM 1
#else
#define USE_TDM 0
#endif

DEV u16 f2bf(float f) {
    u32 u = __builtin_bit_cast(u32, f);
    u32 r = u + 0x7FFFu + ((u >> 16) & 1u);   // round to nearest even
    return (u16)(r >> 16);
}

DEV v8f wmma_bf16(v16bf a, v16bf b, v8f c) {
    return __builtin_amdgcn_wmma_f32_16x16x32_bf16(
        false, a, false, b, (short)0, c, false, false);
}

// A fragment (16x32 bf16): lane<16 -> row=lane, K chunk kb..kb+15 contiguous
DEV v16bf ld_fragA(const u16* base, int strideU, int row, int kb) {
    const v4u* p0 = (const v4u*)(base + (size_t)row * strideU + kb);
    const v4u* p1 = (const v4u*)(base + (size_t)row * strideU + kb + 8);
    FragU f{p0[0], p1[0]};
    return __builtin_bit_cast(v16bf, f);
}

// B fragment (32x16 bf16) from LDS staged [n][k]: 16 contiguous bf16 per lane
DEV v16bf ld_fragB(const u16* p) {
    const v4u* q = (const v4u*)p;
    FragU f{q[0], q[1]};
    return __builtin_bit_cast(v16bf, f);
}

#if USE_TDM
// Issue one TDM 2D tile load: [128 rows x 64 bf16] from row-major [nrows, K]
// tensor at (row0, k0) into LDS at lds_addr. See CDNA5 ISA D# layout.
// amdgpu-toolchain (clang-23) 6-arg builtin form: groups 2/3 + extra int32x8
// are zero for a 2-D tile.
DEV void tdm_tile_load(u32 lds_addr, const u16* gbase, u32 row0, u32 k0,
                       u32 K, u32 nrows) {
    u64 ga = (u64)(size_t)gbase + ((u64)row0 * K + k0) * 2ull;
    v4u g0 = { 1u,                                   // count=1, no gather
               lds_addr,                              // LDS byte address
               (u32)ga,                               // global_addr[31:0]
               ((u32)(ga >> 32) & 0x01FFFFFFu) | (2u << 30) }; // addr[56:32]|type=2
    u64 s0 = (u64)K;                                  // tensor_dim0_stride (elems)
    v8i g1 = { (int)(1u << 16),                       // data_size=1 (2 bytes)
               (int)((K & 0xFFFFu) << 16),            // tensor_dim0[15:0]
               (int)((K >> 16) | ((nrows & 0xFFFFu) << 16)),
               (int)((nrows >> 16) | (64u << 16)),    // tile_dim0 = 64
               (int)(128u),                           // tile_dim1 = 128
               (int)(u32)(s0 & 0xFFFFFFFFu),
               (int)(u32)((s0 >> 32) & 0xFFFFu),
               0 };
    v4i z4 = {0, 0, 0, 0};
    v8i z8 = {0, 0, 0, 0, 0, 0, 0, 0};
    __builtin_amdgcn_tensor_load_to_lds(g0, g1, z4, z4, z8, 0);
}
#endif

// ---------------------------------------------------------------------------
// Elementwise helpers
// ---------------------------------------------------------------------------

__global__ void cvt_f32_bf16(const float* __restrict__ in, u16* __restrict__ out,
                             size_t count) {
    size_t i = (size_t)blockIdx.x * blockDim.x + threadIdx.x;
    if (i < count) out[i] = f2bf(in[i]);
}

// [K,N] f32 -> [N,K] bf16 per layer (blockIdx.z), LDS 32x32 tile transpose
__global__ __launch_bounds__(256)
void transpose_cvt(const float* __restrict__ in, u16* __restrict__ out,
                   int K, int N) {
    __shared__ float t[32][33];
    const size_t lstride = (size_t)K * N;
    const float* src = in + (size_t)blockIdx.z * lstride;
    u16* dst = out + (size_t)blockIdx.z * lstride;
    int k0 = blockIdx.x * 32, n0 = blockIdx.y * 32;
    int i = threadIdx.x >> 5, j = threadIdx.x & 31;
#pragma unroll
    for (int r = 0; r < 4; r++)
        t[i + r * 8][j] = src[(size_t)(k0 + i + r * 8) * N + n0 + j];
    __syncthreads();
#pragma unroll
    for (int r = 0; r < 4; r++)
        dst[(size_t)(n0 + i + r * 8) * K + k0 + j] = f2bf(t[j][i + r * 8]);
}

// pos[n,d] = sph_pos[n,0]*pos_w[0,d] + sph_pos[n,1]*pos_w[1,d] + pos_b[d]
__global__ void pos_kernel(const float* __restrict__ sph_pos,
                           const float* __restrict__ pos_w,
                           const float* __restrict__ pos_b,
                           float* __restrict__ pos) {
    int i = blockIdx.x * blockDim.x + threadIdx.x;   // 128*1024
    int n = i >> 10, d = i & 1023;
    pos[i] = sph_pos[n * 2 + 0] * pos_w[d] +
             sph_pos[n * 2 + 1] * pos_w[1024 + d] + pos_b[d];
}

// A_patch[b*128 + gh*16+gw][c*1024 + p*32 + q] = bf16(img[b,c,gh*32+p,gw*32+q])
__global__ void patch_kernel(const float* __restrict__ img,
                             u16* __restrict__ ap) {
    size_t i = (size_t)blockIdx.x * blockDim.x + threadIdx.x; // 4096*3072
    int row = (int)(i / 3072u);
    int k = (int)(i - (size_t)row * 3072u);
    int b = row >> 7, t = row & 127;
    int gh = t >> 4, gw = t & 15;
    int c = k >> 10, pq = k & 1023;
    int p = pq >> 5, q = pq & 31;
    size_t src = (((size_t)(b * 3 + c) * 256) + gh * 32 + p) * 512 + gw * 32 + q;
    ap[i] = f2bf(img[src]);
}

// ---------------------------------------------------------------------------
// bf16 WMMA GEMM: C[M,N] = A[M,K] x B[N,K]^T (+bias)(+res), fp32 out.
// Block tile 128x128, BK=64, 256 threads = 8 waves, wave tile 32x64.
// LDS double-buffered; tiles staged by the Tensor Data Mover (wave 0 issues,
// TENSORcnt gates, barrier publishes). M%128==0, N%128==0, K%64==0.
// ---------------------------------------------------------------------------
template <bool HAS_BIAS, bool HAS_RES>
__global__ __launch_bounds__(256)
void gemm_bf16(const u16* __restrict__ A, const u16* __restrict__ Bt,
               const float* __restrict__ bias, const float* __restrict__ res,
               float* __restrict__ C, int M, int N, int K) {
    __shared__ alignas(16) u16 lA[2][128 * 64];
    __shared__ alignas(16) u16 lB[2][128 * 64];

    const int tid = threadIdx.x;
    const int lane = tid & 31, w = tid >> 5;
    const int wm = (w >> 1) * 32, wn = (w & 1) * 64;
    const int m0 = blockIdx.y * 128, n0 = blockIdx.x * 128;
    const int nb = K >> 6;

    v8f acc[2][4];
#pragma unroll
    for (int i = 0; i < 2; i++)
#pragma unroll
        for (int j = 0; j < 4; j++) acc[i][j] = v8f{0, 0, 0, 0, 0, 0, 0, 0};

#if USE_TDM
    if (w == 0) {   // DMA wave: prologue loads for buffer 0
        tdm_tile_load((u32)(size_t)&lA[0][0], A,  (u32)m0, 0u, (u32)K, (u32)M);
        tdm_tile_load((u32)(size_t)&lB[0][0], Bt, (u32)n0, 0u, (u32)K, (u32)N);
    }
#endif
    for (int ib = 0; ib < nb; ib++) {
        const int buf = ib & 1;
#if USE_TDM
        if (w == 0) {
            if (ib + 1 < nb) {
                tdm_tile_load((u32)(size_t)&lA[buf ^ 1][0], A,  (u32)m0,
                              (u32)((ib + 1) << 6), (u32)K, (u32)M);
                tdm_tile_load((u32)(size_t)&lB[buf ^ 1][0], Bt, (u32)n0,
                              (u32)((ib + 1) << 6), (u32)K, (u32)N);
                __builtin_amdgcn_s_wait_tensorcnt(2);
            } else {
                __builtin_amdgcn_s_wait_tensorcnt(0);
            }
        }
        __syncthreads();   // current buffer visible to all waves
#else
        {   // fallback: all-thread staging (128 rows x 64 bf16 per tile)
            const int r = tid >> 1, hf = (tid & 1) * 32;
            const int k0 = ib << 6;
            const v4u* sa = (const v4u*)(A  + (size_t)(m0 + r) * K + k0 + hf);
            const v4u* sb = (const v4u*)(Bt + (size_t)(n0 + r) * K + k0 + hf);
            v4u* da = (v4u*)(lA[buf] + r * 64 + hf);
            v4u* db = (v4u*)(lB[buf] + r * 64 + hf);
#pragma unroll
            for (int i = 0; i < 4; i++) { da[i] = sa[i]; db[i] = sb[i]; }
        }
        __syncthreads();
#endif
#pragma unroll
        for (int sub = 0; sub < 64; sub += 32) {
            v16bf af[2], bfr[4];
#pragma unroll
            for (int mt = 0; mt < 2; mt++)
                af[mt] = ld_fragA(lA[buf], 64, wm + mt * 16 + (lane & 15),
                                  sub + (lane >> 4) * 16);
#pragma unroll
            for (int nt = 0; nt < 4; nt++)
                bfr[nt] = ld_fragB(lB[buf] +
                                   (wn + nt * 16 + (lane & 15)) * 64 +
                                   sub + (lane >> 4) * 16);
#pragma unroll
            for (int mt = 0; mt < 2; mt++)
#pragma unroll
                for (int nt = 0; nt < 4; nt++)
                    acc[mt][nt] = wmma_bf16(af[mt], bfr[nt], acc[mt][nt]);
        }
        __syncthreads();   // done reading buf before it is overwritten
    }

    // D layout: vgpr rr -> row = rr + (lane>=16 ? 8 : 0), col = lane&15
#pragma unroll
    for (int mt = 0; mt < 2; mt++)
#pragma unroll
        for (int nt = 0; nt < 4; nt++)
#pragma unroll
            for (int rr = 0; rr < 8; rr++) {
                int row = m0 + wm + mt * 16 + rr + (lane >> 4) * 8;
                int col = n0 + wn + nt * 16 + (lane & 15);
                float v = acc[mt][nt][rr];
                if (HAS_BIAS) v += bias[col];
                if (HAS_RES)  v += res[(size_t)row * N + col];
                C[(size_t)row * N + col] = v;
            }
}

// ---------------------------------------------------------------------------
// LayerNorm over D cols; optional pos add (row%128); bf16 and/or f32 out
// ---------------------------------------------------------------------------
__global__ __launch_bounds__(256)
void ln_kernel(const float* __restrict__ in, int D,
               const float* __restrict__ g, const float* __restrict__ b,
               const float* __restrict__ pos,
               u16* __restrict__ outb, float* __restrict__ outf) {
    __shared__ float s1[256], s2[256];
    const int row = blockIdx.x, tid = threadIdx.x;
    const float* x = in + (size_t)row * D;
    float a1 = 0.f, a2 = 0.f;
    for (int c = tid; c < D; c += 256) { float v = x[c]; a1 += v; a2 += v * v; }
    s1[tid] = a1; s2[tid] = a2; __syncthreads();
    for (int s = 128; s > 0; s >>= 1) {
        if (tid < s) { s1[tid] += s1[tid + s]; s2[tid] += s2[tid + s]; }
        __syncthreads();
    }
    float mu = s1[0] / D;
    float var = s2[0] / D - mu * mu;
    float rs = rsqrtf(var + 1e-5f);
    for (int c = tid; c < D; c += 256) {
        float v = (x[c] - mu) * rs * g[c] + b[c];
        if (pos)  v += pos[(size_t)(row & 127) * D + c];
        if (outf) outf[(size_t)row * D + c] = v;
        if (outb) outb[(size_t)row * D + c] = f2bf(v);
    }
}

// Final LN over 3072 + reassemble: out[b,c,gh*32+p1,gw*32+p2], k=(p1,p2,c)
__global__ __launch_bounds__(256)
void ln_scatter_kernel(const float* __restrict__ in,
                       const float* __restrict__ g, const float* __restrict__ b,
                       float* __restrict__ out) {
    __shared__ float s1[256], s2[256];
    const int row = blockIdx.x, tid = threadIdx.x;   // row = b*128 + gh*16+gw
    const int D = 3072;
    const float* x = in + (size_t)row * D;
    float a1 = 0.f, a2 = 0.f;
    for (int c = tid; c < D; c += 256) { float v = x[c]; a1 += v; a2 += v * v; }
    s1[tid] = a1; s2[tid] = a2; __syncthreads();
    for (int s = 128; s > 0; s >>= 1) {
        if (tid < s) { s1[tid] += s1[tid + s]; s2[tid] += s2[tid + s]; }
        __syncthreads();
    }
    float mu = s1[0] / D;
    float rs = rsqrtf(s2[0] / D - mu * mu + 1e-5f);
    int bb = row >> 7, t = row & 127, gh = t >> 4, gw = t & 15;
    for (int k = tid; k < D; k += 256) {
        float v = (x[k] - mu) * rs * g[k] + b[k];
        int p1 = k / 96, rem = k - p1 * 96, p2 = rem / 3, c = rem - p2 * 3;
        size_t dst = (((size_t)(bb * 3 + c) * 256) + gh * 32 + p1) * 512 +
                     gw * 32 + p2;
        out[dst] = v;
    }
}

// ---------------------------------------------------------------------------
// Attention: dots = (q k^T) * 0.125 * (1 + sph_dist)  per (b,h)
// ---------------------------------------------------------------------------
__global__ __launch_bounds__(256)
void attn_dots_kernel(const float* __restrict__ qkv,
                      const float* __restrict__ sph_dist,
                      float* __restrict__ dots) {
    __shared__ alignas(16) u16 lQ[128 * 64];
    __shared__ alignas(16) u16 lK[128 * 64];
    const int bh = blockIdx.x, b = bh >> 4, hh = bh & 15;
    const int tid = threadIdx.x, lane = tid & 31, w = tid >> 5;
    {
        int n = tid >> 1, dh = (tid & 1) * 32;
        const v4f* qs = (const v4f*)(qkv + ((size_t)(b * 128 + n)) * 3072 +
                                     hh * 64 + dh);
        const v4f* ks = (const v4f*)((const float*)qs + 1024);
#pragma unroll
        for (int i = 0; i < 8; i++) {
            v4f q4 = qs[i], k4 = ks[i];
#pragma unroll
            for (int j = 0; j < 4; j++) {
                lQ[n * 64 + dh + i * 4 + j] = f2bf(q4[j]);
                lK[n * 64 + dh + i * 4 + j] = f2bf(k4[j]);
            }
        }
    }
    __syncthreads();
    v8f acc[8];
#pragma unroll
    for (int i = 0; i < 8; i++) acc[i] = v8f{0, 0, 0, 0, 0, 0, 0, 0};
#pragma unroll
    for (int kk = 0; kk < 64; kk += 32) {
        v16bf a = ld_fragA(lQ, 64, w * 16 + (lane & 15), kk + (lane >> 4) * 16);
#pragma unroll
        for (int nt = 0; nt < 8; nt++) {
            v16bf bb = ld_fragB(lK + (nt * 16 + (lane & 15)) * 64 + kk +
                                (lane >> 4) * 16);
            acc[nt] = wmma_bf16(a, bb, acc[nt]);
        }
    }
#pragma unroll
    for (int nt = 0; nt < 8; nt++)
#pragma unroll
        for (int rr = 0; rr < 8; rr++) {
            int i = w * 16 + rr + (lane >> 4) * 8;
            int j = nt * 16 + (lane & 15);
            dots[((size_t)bh * 128 + i) * 128 + j] =
                acc[nt][rr] * 0.125f * (1.0f + sph_dist[i * 128 + j]);
        }
}

// Cross-head standardize (ddof=1) + softmax over j; attn out bf16
__global__ __launch_bounds__(128)
void attn_softmax_kernel(const float* __restrict__ dots, u16* __restrict__ attn) {
    __shared__ float red[128];
    const int bi = blockIdx.x, b = bi >> 7, i = bi & 127, j = threadIdx.x;
    float wv[16];
#pragma unroll
    for (int h = 0; h < 16; h++)
        wv[h] = dots[(((size_t)(b * 16 + h)) * 128 + i) * 128 + j];
    float mu = 0.f;
#pragma unroll
    for (int h = 0; h < 16; h++) mu += wv[h];
    mu *= (1.0f / 16.0f);
    float var = 0.f;
#pragma unroll
    for (int h = 0; h < 16; h++) { float d = wv[h] - mu; var += d * d; }
    float sd = sqrtf(var / 15.0f);
#pragma unroll
    for (int h = 0; h < 16; h++) wv[h] = (wv[h] - mu) / sd;
    for (int h = 0; h < 16; h++) {
        red[j] = wv[h]; __syncthreads();
        for (int s = 64; s > 0; s >>= 1) {
            if (j < s) red[j] = fmaxf(red[j], red[j + s]);
            __syncthreads();
        }
        float mx = red[0]; __syncthreads();
        float e = __expf(wv[h] - mx);
        red[j] = e; __syncthreads();
        for (int s = 64; s > 0; s >>= 1) {
            if (j < s) red[j] += red[j + s];
            __syncthreads();
        }
        float sm = red[0]; __syncthreads();
        attn[(((size_t)(b * 16 + h)) * 128 + i) * 128 + j] = f2bf(e / sm);
    }
}

// o = attn[128,128] x v[128,64] per (b,h); o out bf16 [b,i,h*64+d]
__global__ __launch_bounds__(256)
void attn_v_kernel(const u16* __restrict__ attn, const float* __restrict__ qkv,
                   u16* __restrict__ o) {
    __shared__ alignas(16) u16 lA[128 * 128];   // 32KB
    __shared__ alignas(16) u16 lV[64 * 128];    // [d][j], 16KB
    const int bh = blockIdx.x, b = bh >> 4, hh = bh & 15;
    const int tid = threadIdx.x, lane = tid & 31, w = tid >> 5;
    {
        int row = tid >> 1, hf = (tid & 1) * 64;
        const v4u* src = (const v4u*)(attn + ((size_t)bh * 128 + row) * 128 + hf);
        v4u* dst = (v4u*)(lA + row * 128 + hf);
#pragma unroll
        for (int i = 0; i < 8; i++) dst[i] = src[i];
    }
    {
        int j = tid >> 1, dh = (tid & 1) * 32;
        const v4f* vs = (const v4f*)(qkv + ((size_t)(b * 128 + j)) * 3072 +
                                     2048 + hh * 64 + dh);
#pragma unroll
        for (int i = 0; i < 8; i++) {
            v4f v4 = vs[i];
#pragma unroll
            for (int jj = 0; jj < 4; jj++)
                lV[(dh + i * 4 + jj) * 128 + j] = f2bf(v4[jj]);
        }
    }
    __syncthreads();
    v8f acc[4];
#pragma unroll
    for (int i = 0; i < 4; i++) acc[i] = v8f{0, 0, 0, 0, 0, 0, 0, 0};
#pragma unroll
    for (int kk = 0; kk < 128; kk += 32) {
        v16bf a = ld_fragA(lA, 128, w * 16 + (lane & 15), kk + (lane >> 4) * 16);
#pragma unroll
        for (int nt = 0; nt < 4; nt++) {
            v16bf bb = ld_fragB(lV + (nt * 16 + (lane & 15)) * 128 + kk +
                                (lane >> 4) * 16);
            acc[nt] = wmma_bf16(a, bb, acc[nt]);
        }
    }
#pragma unroll
    for (int nt = 0; nt < 4; nt++)
#pragma unroll
        for (int rr = 0; rr < 8; rr++) {
            int i = w * 16 + rr + (lane >> 4) * 8;
            int d = nt * 16 + (lane & 15);
            o[((size_t)(b * 128 + i)) * 1024 + hh * 64 + d] = f2bf(acc[nt][rr]);
        }
}

// ---------------------------------------------------------------------------
// Host orchestration
// ---------------------------------------------------------------------------
extern "C" void kernel_launch(void* const* d_in, const int* in_sizes, int n_in,
                              void* d_out, int out_size, void* d_ws,
                              size_t ws_size, hipStream_t stream) {
    (void)in_sizes; (void)n_in; (void)out_size; (void)ws_size;
    const float* img      = (const float*)d_in[0];
    const float* sph_pos  = (const float*)d_in[1];
    const float* sph_dist = (const float*)d_in[2];
    const float* conv_w   = (const float*)d_in[3];
    const float* conv_b   = (const float*)d_in[4];
    const float* nl_ln1_g = (const float*)d_in[5];
    const float* nl_ln1_b = (const float*)d_in[6];
    const float* nl_w     = (const float*)d_in[7];
    const float* nl_b     = (const float*)d_in[8];
    const float* nl_ln2_g = (const float*)d_in[9];
    const float* nl_ln2_b = (const float*)d_in[10];
    const float* pos_w    = (const float*)d_in[11];
    const float* pos_b    = (const float*)d_in[12];
    const float* ln_g     = (const float*)d_in[13];
    const float* ln_b     = (const float*)d_in[14];
    const float* wqkv     = (const float*)d_in[15];
    const float* wo       = (const float*)d_in[16];
    const float* wo_b     = (const float*)d_in[17];
    const float* tr_ln_g  = (const float*)d_in[18];
    const float* tr_ln_b  = (const float*)d_in[19];
    const float* exp_w    = (const float*)d_in[20];
    const float* exp_b    = (const float*)d_in[21];
    const float* out_ln_g = (const float*)d_in[22];
    const float* out_ln_b = (const float*)d_in[23];
    float* out = (float*)d_out;

    const int M = 4096;   // B*N tokens

    // ---- workspace bump allocation (256B aligned) ----
    char* base = (char*)d_ws;
    size_t off = 0;
    auto alloc = [&](size_t bytes) -> char* {
        off = (off + 255) & ~(size_t)255;
        char* p = base + off;
        off += bytes;
        return p;
    };
    u16*   wconv_bf = (u16*)alloc((size_t)1024 * 3072 * 2);       // [N,K] direct
    u16*   wnl_bf   = (u16*)alloc((size_t)1024 * 1024 * 2);       // [N,K] transposed
    u16*   wqkv_bf  = (u16*)alloc((size_t)12 * 1024 * 3072 * 2);  // [N,K] transposed
    u16*   wo_bf    = (u16*)alloc((size_t)12 * 1024 * 1024 * 2);  // [N,K] transposed
    u16*   wexp_bf  = (u16*)alloc((size_t)1024 * 3072 * 2);       // [N,K] transposed
    float* xbuf     = (float*)alloc((size_t)M * 1024 * 4);
    u16*   hbuf     = (u16*)alloc((size_t)M * 1024 * 2);
    float* posbuf   = (float*)alloc((size_t)128 * 1024 * 4);
    float* qkvbuf   = (float*)alloc((size_t)M * 3072 * 4);        // also 'e'
    float* dotsbuf  = (float*)alloc((size_t)512 * 128 * 128 * 4);
    char*  shared25 = alloc((size_t)M * 3072 * 2);                // 25.2MB pool
    u16*   apatch   = (u16*)shared25;                 // prologue
    float* tmpbuf   = (float*)shared25;               // prologue (after apatch)
    u16*   attnbuf  = (u16*)shared25;                 // layer loop
    u16*   obuf     = (u16*)(shared25 + (size_t)512 * 128 * 128 * 2);

    // ---- weights -> bf16 ([N,K] everywhere) ----
    cvt_f32_bf16<<<dim3((unsigned)(((size_t)1024 * 3072 + 255) / 256)), 256, 0,
                   stream>>>(conv_w, wconv_bf, (size_t)1024 * 3072);
    transpose_cvt<<<dim3(32, 32, 1),  256, 0, stream>>>(nl_w,  wnl_bf,  1024, 1024);
    transpose_cvt<<<dim3(32, 96, 12), 256, 0, stream>>>(wqkv,  wqkv_bf, 1024, 3072);
    transpose_cvt<<<dim3(32, 32, 12), 256, 0, stream>>>(wo,    wo_bf,   1024, 1024);
    transpose_cvt<<<dim3(32, 96, 1),  256, 0, stream>>>(exp_w, wexp_bf, 1024, 3072);

    pos_kernel<<<dim3(128 * 1024 / 256), 256, 0, stream>>>(sph_pos, pos_w,
                                                           pos_b, posbuf);
    patch_kernel<<<dim3((unsigned)((size_t)M * 3072 / 256)), 256, 0, stream>>>(
        img, apatch);

    // patch embed: x = A_patch[4096,3072] x conv_w[1024,3072]^T + conv_b
    gemm_bf16<true, false><<<dim3(1024 / 128, M / 128), 256, 0, stream>>>(
        apatch, wconv_bf, conv_b, nullptr, xbuf, M, 1024, 3072);

    // norm_linear
    ln_kernel<<<dim3(M), 256, 0, stream>>>(xbuf, 1024, nl_ln1_g, nl_ln1_b,
                                           nullptr, hbuf, nullptr);
    gemm_bf16<true, false><<<dim3(1024 / 128, M / 128), 256, 0, stream>>>(
        hbuf, wnl_bf, nl_b, nullptr, tmpbuf, M, 1024, 1024);
    ln_kernel<<<dim3(M), 256, 0, stream>>>(tmpbuf, 1024, nl_ln2_g, nl_ln2_b,
                                           posbuf, nullptr, xbuf);

    // transformer layers
    for (int l = 0; l < 12; l++) {
        ln_kernel<<<dim3(M), 256, 0, stream>>>(
            xbuf, 1024, ln_g + (size_t)l * 1024, ln_b + (size_t)l * 1024,
            nullptr, hbuf, nullptr);
        gemm_bf16<false, false>
            <<<dim3(3072 / 128, M / 128), 256, 0, stream>>>(
                hbuf, wqkv_bf + (size_t)l * 1024 * 3072, nullptr, nullptr,
                qkvbuf, M, 3072, 1024);
        attn_dots_kernel<<<dim3(512), 256, 0, stream>>>(qkvbuf, sph_dist,
                                                        dotsbuf);
        attn_softmax_kernel<<<dim3(M), 128, 0, stream>>>(dotsbuf, attnbuf);
        attn_v_kernel<<<dim3(512), 256, 0, stream>>>(attnbuf, qkvbuf, obuf);
        gemm_bf16<true, true>
            <<<dim3(1024 / 128, M / 128), 256, 0, stream>>>(
                obuf, wo_bf + (size_t)l * 1024 * 1024,
                wo_b + (size_t)l * 1024, xbuf, xbuf, M, 1024, 1024);
    }

    // final LN + expand + output LN/reassemble
    ln_kernel<<<dim3(M), 256, 0, stream>>>(xbuf, 1024, tr_ln_g, tr_ln_b,
                                           nullptr, hbuf, nullptr);
    gemm_bf16<true, false><<<dim3(3072 / 128, M / 128), 256, 0, stream>>>(
        hbuf, wexp_bf, exp_b, nullptr, qkvbuf /* e */, M, 3072, 1024);
    ln_scatter_kernel<<<dim3(M), 256, 0, stream>>>(qkvbuf, out_ln_g, out_ln_b,
                                                   out);
}
